// SchNETInteractionModule_1709396984150
// MI455X (gfx1250) — compile-verified, matrix operand order checked by reading, not run on hardware
//
#include <hip/hip_runtime.h>
#include <hip/hip_bf16.h>

typedef __attribute__((ext_vector_type(16))) __bf16 v16bf;
typedef __attribute__((ext_vector_type(8)))  float  v8f;

#define N_ATOMS 50000
#define N_PAIRS 1600000
#define FDIM 128
#define RBF 20
#define LN2 0.69314718056f

__device__ __forceinline__ v8f wmma_bf16(v16bf a, v16bf b, v8f c) {
    return __builtin_amdgcn_wmma_f32_16x16x32_bf16(
        /*neg_a=*/false, a, /*neg_b=*/false, b,
        /*c_mod=*/(short)0, c, /*reuse_a=*/false, /*reuse_b=*/false);
}

__device__ __forceinline__ float ssp(float x) {
    // shifted softplus: softplus(x) - ln2, numerically stable
    return fmaxf(x, 0.0f) + __logf(1.0f + __expf(-fabsf(x))) - LN2;
}

// ---------------- zero the output accumulator ----------------
__global__ void zero_kernel(float* __restrict__ p, int n) {
    int i = blockIdx.x * blockDim.x + threadIdx.x;
    if (i < n) p[i] = 0.0f;
}

// ---- convert all weights to bf16, transposed to [N][K] (B-fragment friendly) ----
__global__ void prep_weights(const float* __restrict__ W_in,
                             const float* __restrict__ W_f1,
                             const float* __restrict__ W_f2,
                             const float* __restrict__ W_o1,
                             const float* __restrict__ W_o2,
                             __bf16* __restrict__ Wt_in,
                             __bf16* __restrict__ Wt_f1,
                             __bf16* __restrict__ Wt_f2,
                             __bf16* __restrict__ Wt_o1,
                             __bf16* __restrict__ Wt_o2) {
    int idx = blockIdx.x * blockDim.x + threadIdx.x;
    if (idx >= FDIM * FDIM) return;
    int n = idx >> 7;      // output column
    int k = idx & 127;     // input row
    Wt_in[n * FDIM + k] = (__bf16)W_in[k * FDIM + n];
    Wt_f2[n * FDIM + k] = (__bf16)W_f2[k * FDIM + n];
    Wt_o1[n * FDIM + k] = (__bf16)W_o1[k * FDIM + n];
    Wt_o2[n * FDIM + k] = (__bf16)W_o2[k * FDIM + n];
    if (k < 32)  // K padded 20 -> 32 so layer-1 is a single 16x16x32 WMMA
        Wt_f1[n * 32 + k] = (k < RBF) ? (__bf16)W_f1[k * FDIM + n] : (__bf16)0.0f;
}

// ---------------- x = emb @ W_in  (one wave per 16-row tile) ----------------
__global__ void __launch_bounds__(32)
xproj_kernel(const float* __restrict__ emb, const __bf16* __restrict__ Wt_in,
             float* __restrict__ x) {
    int lane = threadIdx.x, half = lane >> 4, l16 = lane & 15;
    int row0 = blockIdx.x * 16;
    // A fragments: lane holds row l16, K chunk ks*32 + half*16 .. +16
    v16bf a[4];
    for (int ks = 0; ks < 4; ++ks) {
        const float* src = emb + (row0 + l16) * FDIM + ks * 32 + half * 16;
        v16bf av;
        for (int t = 0; t < 16; ++t) av[t] = (__bf16)src[t];
        a[ks] = av;
    }
    for (int nt = 0; nt < 8; ++nt) {
        int n = nt * 16 + l16;
        v8f c = {};
        for (int ks = 0; ks < 4; ++ks) {
            v16bf b = *(const v16bf*)(Wt_in + n * FDIM + ks * 32 + half * 16);
            c = wmma_bf16(a[ks], b, c);
        }
        // D layout: lane -> column n, rows v + 8*half
        for (int v = 0; v < 8; ++v)
            x[(row0 + v + 8 * half) * FDIM + n] = c[v];
    }
}

// ------- fused: filter MLP -> cutoff -> gather x[idx_j] -> scatter-add out[idx_i] -------
__global__ void __launch_bounds__(32)
pair_kernel(const float* __restrict__ f_ij, const float* __restrict__ cutoff,
            const int* __restrict__ pair_idx, const float* __restrict__ x,
            const __bf16* __restrict__ Wt_f1, const float* __restrict__ b_f1,
            const __bf16* __restrict__ Wt_f2, const float* __restrict__ b_f2,
            float* __restrict__ out) {
    __shared__ __bf16 lds_h[16 * FDIM];   // activations, A-layout staging
    __shared__ float  lds_cut[16];
    __shared__ int    lds_i[16], lds_j[16];

    int lane = threadIdx.x, half = lane >> 4, l16 = lane & 15;
    int p0 = blockIdx.x * 16;

    if (lane < 16) {
        lds_cut[lane] = cutoff[p0 + lane];
        lds_i[lane]   = pair_idx[p0 + lane];            // idx_i
        lds_j[lane]   = pair_idx[N_PAIRS + p0 + lane];  // idx_j
    }
    __syncthreads();

    // A1: f_ij row (pair p0+l16), K padded 20->32
    v16bf a1;
    {
        const float* src = f_ij + (p0 + l16) * RBF;
        int k0 = half * 16;
        for (int t = 0; t < 16; ++t) {
            int k = k0 + t;
            a1[t] = (__bf16)((k < RBF) ? src[k] : 0.0f);
        }
    }

    // layer 1: single WMMA per N-tile (K=32), + bias + shifted softplus -> LDS (bf16)
    for (int nt = 0; nt < 8; ++nt) {
        int n = nt * 16 + l16;
        v16bf b = *(const v16bf*)(Wt_f1 + n * 32 + half * 16);
        v8f c = {};
        c = wmma_bf16(a1, b, c);
        float bias = b_f1[n];
        for (int v = 0; v < 8; ++v) {
            float h = ssp(c[v] + bias);
            lds_h[(v + 8 * half) * FDIM + n] = (__bf16)h;
        }
    }
    __syncthreads();

    // layer 2 A fragments from LDS (row l16, contiguous K chunks)
    v16bf a2[4];
    for (int ks = 0; ks < 4; ++ks)
        a2[ks] = *(const v16bf*)(lds_h + l16 * FDIM + ks * 32 + half * 16);

    for (int nt = 0; nt < 8; ++nt) {
        int n = nt * 16 + l16;
        v8f c = {};
        for (int ks = 0; ks < 4; ++ks) {
            v16bf b = *(const v16bf*)(Wt_f2 + n * FDIM + ks * 32 + half * 16);
            c = wmma_bf16(a2[ks], b, c);
        }
        float bias = b_f2[n];
        // cutoff-scale, gather neighbor feature, scatter-add (L2-resident)
        for (int v = 0; v < 8; ++v) {
            int r = v + 8 * half;                 // local pair row
            float w = (c[v] + bias) * lds_cut[r];
            float val = x[lds_j[r] * FDIM + n] * w;
            atomicAdd(&out[lds_i[r] * FDIM + n], val);
        }
    }
}

// ---------------- output MLP, in-place on d_out (per-tile rows only) ----------------
__global__ void __launch_bounds__(32)
outmlp_kernel(const __bf16* __restrict__ Wt_o1, const float* __restrict__ b_o1,
              const __bf16* __restrict__ Wt_o2, const float* __restrict__ b_o2,
              float* __restrict__ out) {
    __shared__ __bf16 lds_h[16 * FDIM];
    int lane = threadIdx.x, half = lane >> 4, l16 = lane & 15;
    int row0 = blockIdx.x * 16;

    // preload ALL reads of this tile before any writes (in-place safety)
    v16bf a[4];
    for (int ks = 0; ks < 4; ++ks) {
        const float* src = out + (row0 + l16) * FDIM + ks * 32 + half * 16;
        v16bf av;
        for (int t = 0; t < 16; ++t) av[t] = (__bf16)src[t];
        a[ks] = av;
    }

    for (int nt = 0; nt < 8; ++nt) {
        int n = nt * 16 + l16;
        v8f c = {};
        for (int ks = 0; ks < 4; ++ks) {
            v16bf b = *(const v16bf*)(Wt_o1 + n * FDIM + ks * 32 + half * 16);
            c = wmma_bf16(a[ks], b, c);
        }
        float bias = b_o1[n];
        for (int v = 0; v < 8; ++v)
            lds_h[(v + 8 * half) * FDIM + n] = (__bf16)ssp(c[v] + bias);
    }
    __syncthreads();

    v16bf a2[4];
    for (int ks = 0; ks < 4; ++ks)
        a2[ks] = *(const v16bf*)(lds_h + l16 * FDIM + ks * 32 + half * 16);

    for (int nt = 0; nt < 8; ++nt) {
        int n = nt * 16 + l16;
        v8f c = {};
        for (int ks = 0; ks < 4; ++ks) {
            v16bf b = *(const v16bf*)(Wt_o2 + n * FDIM + ks * 32 + half * 16);
            c = wmma_bf16(a2[ks], b, c);
        }
        float bias = b_o2[n];
        for (int v = 0; v < 8; ++v)
            out[(row0 + v + 8 * half) * FDIM + n] = c[v] + bias;
    }
}

extern "C" void kernel_launch(void* const* d_in, const int* in_sizes, int n_in,
                              void* d_out, int out_size, void* d_ws, size_t ws_size,
                              hipStream_t stream) {
    const float* emb     = (const float*)d_in[0];
    const int*   pidx    = (const int*)  d_in[1];
    const float* f_ij    = (const float*)d_in[2];
    const float* cut     = (const float*)d_in[3];
    const float* W_in    = (const float*)d_in[4];
    const float* W_f1    = (const float*)d_in[5];
    const float* b_f1    = (const float*)d_in[6];
    const float* W_f2    = (const float*)d_in[7];
    const float* b_f2    = (const float*)d_in[8];
    const float* W_o1    = (const float*)d_in[9];
    const float* b_o1    = (const float*)d_in[10];
    const float* W_o2    = (const float*)d_in[11];
    const float* b_o2    = (const float*)d_in[12];
    float* out = (float*)d_out;

    // workspace carve (all 256B aligned by construction)
    char* ws = (char*)d_ws;
    __bf16* Wt_f1 = (__bf16*)ws;  ws += 128 * 32  * sizeof(__bf16);   // 8 KB
    __bf16* Wt_f2 = (__bf16*)ws;  ws += 128 * 128 * sizeof(__bf16);   // 32 KB
    __bf16* Wt_in = (__bf16*)ws;  ws += 128 * 128 * sizeof(__bf16);
    __bf16* Wt_o1 = (__bf16*)ws;  ws += 128 * 128 * sizeof(__bf16);
    __bf16* Wt_o2 = (__bf16*)ws;  ws += 128 * 128 * sizeof(__bf16);
    float*  x     = (float*)ws;   // 50000*128*4 = 25.6 MB (L2-resident)

    // 1) zero accumulator (d_out doubles as segment_sum accumulator)
    zero_kernel<<<(N_ATOMS * FDIM + 255) / 256, 256, 0, stream>>>(out, N_ATOMS * FDIM);
    // 2) weight convert+transpose to bf16 [N][K]
    prep_weights<<<(FDIM * FDIM + 255) / 256, 256, 0, stream>>>(
        W_in, W_f1, W_f2, W_o1, W_o2, Wt_in, Wt_f1, Wt_f2, Wt_o1, Wt_o2);
    // 3) x = emb @ W_in
    xproj_kernel<<<N_ATOMS / 16, 32, 0, stream>>>(emb, Wt_in, x);
    // 4) fused filter-MLP + gather + scatter-add
    pair_kernel<<<N_PAIRS / 16, 32, 0, stream>>>(
        f_ij, cut, pidx, x, Wt_f1, b_f1, Wt_f2, b_f2, out);
    // 5) output MLP in-place
    outmlp_kernel<<<N_ATOMS / 16, 32, 0, stream>>>(Wt_o1, b_o1, Wt_o2, b_o2, out);
}
